// ContextualAttention_Enhance_58291296141938
// MI455X (gfx1250) — matrix operand order repaired and checked by reading, not compiled, not run on hardware
//
#include <hip/hip_runtime.h>
#include <hip/hip_bf16.h>

typedef __attribute__((ext_vector_type(16))) __bf16 v16bf;
typedef __attribute__((ext_vector_type(8)))  float  v8f;

#define B_      4
#define CIN     64
#define HH      128
#define WW      128
#define HP      134       // H + 2*pad (pad=3)
#define WP      134
#define WPV     144       // padded row stride for V map (16B-aligned chunks)
#define PADP    3
#define NQ      1024      // 32x32 query grid (stride 4)
#define DQ      784       // 16ch * 49 patch positions
#define KSTEPS  25        // 800 padded dims / 32
#define SCALE_F 10.0f

union ABu { v16bf v; unsigned u[8]; __bf16 h[16]; };

static __device__ __forceinline__ int imin(int a, int b) { return a < b ? a : b; }

// CDNA5 async global->LDS copy (ASYNCcnt-tracked, VFLAT encoding).
static __device__ __forceinline__ void async_ldsb128(unsigned ldsoff, const void* gp) {
  asm volatile("global_load_async_to_lds_b128 %0, %1, off"
               :: "v"(ldsoff), "v"(gp) : "memory");
}
#define WAIT_ASYNC(n) asm volatile("s_wait_asynccnt " #n ::: "memory")

// ---------------------------------------------------------------------------
// Kernel 1: 1x1 convs -> padded bf16 feature maps.
//   o1 (queries, g_w)  : [B][HP][WP][16]   channel-last
//   o3 (keys,    ph_w) : [B][HP][WP][16]   channel-last
//   o2 (values,  th_w) : [B][16][HP][WPV]  channel-first, padded stride
// Borders are exact zeros (reference pads post-conv output).
// ---------------------------------------------------------------------------
__global__ void qkv_pack_kernel(const float* __restrict__ x,
                                const float* __restrict__ gw, const float* __restrict__ gb,
                                const float* __restrict__ tw, const float* __restrict__ tb,
                                const float* __restrict__ pw, const float* __restrict__ pb,
                                __bf16* __restrict__ o1, __bf16* __restrict__ o3,
                                __bf16* __restrict__ o2)
{
  const long long tid = (long long)blockIdx.x * 256 + threadIdx.x;
  const long long total = (long long)B_ * HP * WP * 48;
  if (tid >= total) return;
  const int co = (int)(tid % 48);
  long long t = tid / 48;
  const int xx = (int)(t % WP); t /= WP;
  const int yy = (int)(t % HP);
  const int b  = (int)(t / HP);

  float r = 0.0f;
  if (yy >= PADP && yy < HP - PADP && xx >= PADP && xx < WP - PADP) {
    const float* w; float bias;
    if (co < 16)      { w = gw + co * 64;        bias = gb[co]; }
    else if (co < 32) { w = pw + (co - 16) * 64; bias = pb[co - 16]; }
    else              { w = tw + (co - 32) * 64; bias = tb[co - 32]; }
    const float* xi = x + (size_t)b * CIN * (HH * WW)
                        + (size_t)(yy - PADP) * WW + (xx - PADP);
    r = bias;
    #pragma unroll 8
    for (int ci = 0; ci < CIN; ++ci) r += w[ci] * xi[(size_t)ci * (HH * WW)];
  }
  const __bf16 v = (__bf16)r;
  if (co < 16)      o1[(((size_t)b * HP + yy) * WP + xx) * 16 + co] = v;
  else if (co < 32) o3[(((size_t)b * HP + yy) * WP + xx) * 16 + (co - 16)] = v;
  else              o2[((size_t)(b * 16 + (co - 32)) * HP + yy) * WPV + xx] = v;
}

// ---------------------------------------------------------------------------
// Kernel 2: streaming flash-attention, bf16 WMMA + fp32 accum, with
// double-buffered async global->LDS (ASYNCcnt) prefetch of K/V tiles.
// One WG (4 waves) per 16-query tile; 256 key-blocks of 64 keys.
// d-ordering: d = p*16 + c  (p = ph*7+pw), padded to 800 with zeros.
// ---------------------------------------------------------------------------
__global__ __launch_bounds__(128) void attn_kernel(
    const __bf16* __restrict__ qsrc,   // [B][HP][WP][16]
    const __bf16* __restrict__ ksrc,   // [B][HP][WP][16]
    const __bf16* __restrict__ vsrc,   // [B][16][HP][WPV]
    float* __restrict__ Z)             // [B][NQ][DQ]
{
  __shared__ __bf16 Qraw[7 * 68 * 16];        // [ph][x(68)][c]
  __shared__ __bf16 Kraw[2][7 * 72 * 16];     // [buf][ph][x(72)][c] (70 valid)
  __shared__ __bf16 Vraw[2][16 * 7 * 72];     // [buf][c][ph][x(72)] (70 valid)
  __shared__ __bf16 Pbuf[16 * 64];            // probabilities [query][key]
  __shared__ float  wmaxS[4 * 16];
  __shared__ float  wsumS[4 * 16];
  __shared__ uint4  dumpbuf;                  // sink for dummy async lanes

  const int tid  = threadIdx.x;
  const int wid  = tid >> 5;
  const int lane = tid & 31;
  const int half = lane >> 4;
  const int l16  = lane & 15;

  const int g   = blockIdx.x;
  const int b   = g >> 6;
  const int qt  = g & 63;
  const int qy  = qt >> 1;            // all 16 queries share grid row
  const int qx0 = (qt & 1) * 16;
  const int ys  = qy * 4;
  const int xs0 = qx0 * 4;

  const unsigned kraw0 = (unsigned)(size_t)(&Kraw[0][0]);
  const unsigned vraw0 = (unsigned)(size_t)(&Vraw[0][0]);
  const unsigned dumpo = (unsigned)(size_t)(&dumpbuf);

  // ---- load Q raw tile: 7 rows x 68 px x 16 ch (contiguous per row) ----
  {
    const size_t base = ((size_t)b * HP + ys) * WP + xs0;
    for (int idx = tid; idx < 7 * 136; idx += 128) {
      const int ph = idx / 136, ch = idx - ph * 136;
      const uint4* src = (const uint4*)(qsrc + (base + (size_t)ph * WP) * 16) + ch;
      *(uint4*)&Qraw[ph * 68 * 16 + ch * 8] = *src;
    }
  }

  // ---- async prefetch of a key-block tile into LDS buffer `buf` ----
  // Exactly 16 async instructions per wave (8 K + 8 V): dummies use selects,
  // never divergence, so s_wait_asynccnt thresholds are exact.
  auto issue_block = [&](int nb, int buf) {
    const int ky = nb >> 1, x0 = (nb & 1) << 6;
    const size_t kbase = ((size_t)b * HP + ky) * WP + x0;
    #pragma unroll
    for (int it = 0; it < 8; ++it) {               // K: 980 real b128 chunks
      const int idx = tid + it * 128;
      const int ph = idx / 140, ch = idx - ph * 140;
      const bool real = idx < 7 * 140;
      const void* gp = real
          ? (const void*)((const uint4*)(ksrc + (kbase + (size_t)ph * WP) * 16) + ch)
          : (const void*)ksrc;
      const unsigned lo = real
          ? (kraw0 + (unsigned)buf * (7u*72*16*2) + (unsigned)(ph * 2304 + ch * 16))
          : dumpo;
      async_ldsb128(lo, gp);
    }
    #pragma unroll
    for (int it = 0; it < 8; ++it) {               // V: 1008 real b128 chunks
      const int idx = tid + it * 128;
      const int r  = idx / 9, ch = idx - r * 9;
      const int c  = r / 7, ph = r - c * 7;
      const bool real = idx < 112 * 9;
      const void* gp = real
          ? (const void*)(vsrc + ((size_t)(b * 16 + c) * HP + (ky + ph)) * WPV
                          + x0 + ch * 8)
          : (const void*)vsrc;
      const unsigned lo = real
          ? (vraw0 + (unsigned)buf * (16u*7*72*2) + (unsigned)(r * 144 + ch * 16))
          : dumpo;
      async_ldsb128(lo, gp);
    }
  };

  float mrun[8], lrun[8];
  v8f acc[13];
  #pragma unroll
  for (int i = 0; i < 8; ++i) { mrun[i] = -1e30f; lrun[i] = 0.0f; }
  #pragma unroll
  for (int t = 0; t < 13; ++t) {
    #pragma unroll
    for (int i = 0; i < 8; ++i) acc[t][i] = 0.0f;
  }

  const int keyx = wid * 16 + l16;    // this wave's key (x within 64-block)

  issue_block(0, 0);                  // prologue prefetch

  #pragma unroll 1
  for (int nb = 0; nb < 256; ++nb) {
    const int cur = nb & 1;

    __syncthreads();                  // all waves done with buf[cur^1] & P/stats
    if (nb < 255) {
      issue_block(nb + 1, cur ^ 1);
      WAIT_ASYNC(16);                 // previous block's 16 ops complete
    } else {
      WAIT_ASYNC(0);
    }
    __syncthreads();                  // block nb tiles visible to all waves

    const __bf16* Kb = Kraw[cur];
    const __bf16* Vb = Vraw[cur];

    // ---- scores S[16q x 16k] for this wave's keys: 25 bf16 WMMAs ----
    v8f S;
    #pragma unroll
    for (int i = 0; i < 8; ++i) S[i] = 0.0f;

    #pragma unroll
    for (int ks = 0; ks < KSTEPS; ++ks) {
      ABu a;   // A operand: Q, lane=query l16, K-dims contiguous: half -> p=2ks+half, elem s -> c=s
      #pragma unroll
      for (int v = 0; v < 8; ++v) {
        const int p = 2 * ks + half;
        unsigned val = 0u;
        if (p < 49) {
          const int ph = p / 7, pw = p % 7;            // compile-time
          const int c  = 2 * v;
          val = *(const unsigned*)&Qraw[(ph * 68 + l16 * 4 + pw) * 16 + c];
        }
        a.u[v] = val;
      }
      ABu bb;  // B operand: K, lane=key, VGPR j -> dim pair (2j,2j+1)
      {
        const int ro0 = ((2 * ks) / 7) * 72 + ((2 * ks) % 7);
        const int ro1 = ((2 * ks + 1) / 7) * 72 + ((2 * ks + 1) % 7);
        const int ro  = half ? ro1 : ro0;
        const bool ok = (2 * ks + half) < 49;          // false only ks=24,half=1
        #pragma unroll
        for (int j = 0; j < 8; ++j) {
          unsigned val = 0u;
          if (ok) val = *(const unsigned*)&Kb[(ro + keyx) * 16 + 2 * j];
          bb.u[j] = val;
        }
      }
      S = __builtin_amdgcn_wmma_f32_16x16x32_bf16(false, a.v, false, bb.v,
                                                  (short)0, S, false, false);
    }

    // ---- online softmax (row = i + 8*half per C layout) ----
    float s[8], mx[8];
    #pragma unroll
    for (int i = 0; i < 8; ++i) {
      float xv = S[i] * SCALE_F;
      s[i] = xv;
      xv = fmaxf(xv, __shfl_xor(xv, 1, 32));
      xv = fmaxf(xv, __shfl_xor(xv, 2, 32));
      xv = fmaxf(xv, __shfl_xor(xv, 4, 32));
      xv = fmaxf(xv, __shfl_xor(xv, 8, 32));
      mx[i] = xv;
    }
    if (l16 == 0) {
      #pragma unroll
      for (int i = 0; i < 8; ++i) wmaxS[wid * 16 + half * 8 + i] = mx[i];
    }
    __syncthreads();

    float cc[8];
    #pragma unroll
    for (int i = 0; i < 8; ++i) {
      const int row = i + 8 * half;
      float bm = wmaxS[row];
      bm = fmaxf(bm, wmaxS[16 + row]);
      bm = fmaxf(bm, wmaxS[32 + row]);
      bm = fmaxf(bm, wmaxS[48 + row]);
      const float mnew = fmaxf(mrun[i], bm);
      cc[i] = __expf(mrun[i] - mnew);
      mrun[i] = mnew;
    }
    float ps[8];
    #pragma unroll
    for (int i = 0; i < 8; ++i) {
      float p = __expf(s[i] - mrun[i]);
      s[i] = p;
      float xv = p;
      xv += __shfl_xor(xv, 1, 32);
      xv += __shfl_xor(xv, 2, 32);
      xv += __shfl_xor(xv, 4, 32);
      xv += __shfl_xor(xv, 8, 32);
      ps[i] = xv;
    }
    if (l16 == 0) {
      #pragma unroll
      for (int i = 0; i < 8; ++i) wsumS[wid * 16 + half * 8 + i] = ps[i];
    }
    #pragma unroll
    for (int i = 0; i < 8; ++i)
      Pbuf[(i + 8 * half) * 64 + wid * 16 + l16] = (__bf16)s[i];
    __syncthreads();

    #pragma unroll
    for (int i = 0; i < 8; ++i) {
      const int row = i + 8 * half;
      const float bs = wsumS[row] + wsumS[16 + row] + wsumS[32 + row] + wsumS[48 + row];
      lrun[i] = cc[i] * lrun[i] + bs;
    }
    #pragma unroll
    for (int t = 0; t < 13; ++t) {
      #pragma unroll
      for (int i = 0; i < 8; ++i) acc[t][i] *= cc[i];
    }

    // ---- PV: O[16q x 16d] per dim tile (tile tt == patch position p) ----
    ABu pa0, pa1;  // A operand: P, K-dim = key 0..31 / 32..63
    #pragma unroll
    for (int v = 0; v < 8; ++v) {
      const int kb = 16 * (v >> 2) + 8 * half + 2 * (v & 3);
      pa0.u[v] = *(const unsigned*)&Pbuf[l16 * 64 + kb];
      pa1.u[v] = *(const unsigned*)&Pbuf[l16 * 64 + 32 + kb];
    }
    #pragma unroll
    for (int t = 0; t < 13; ++t) {
      const int tt = wid + 4 * t;
      if (tt < 49) {
        const int ph = tt / 7, pw = tt - 7 * ph;
        const int vb_base = (l16 * 7 + ph) * 72 + pw;   // lane = channel c
        ABu vb0, vb1;  // B operand: V, VGPR j -> key pair, from [c][ph][x]
        #pragma unroll
        for (int j = 0; j < 8; ++j) {
          const int kk = 16 * (j >> 2) + 8 * half + 2 * (j & 3);
          vb0.h[2 * j]     = Vb[vb_base + kk];
          vb0.h[2 * j + 1] = Vb[vb_base + kk + 1];
          vb1.h[2 * j]     = Vb[vb_base + 32 + kk];
          vb1.h[2 * j + 1] = Vb[vb_base + 32 + kk + 1];
        }
        v8f A = acc[t];
        A = __builtin_amdgcn_wmma_f32_16x16x32_bf16(false, pa0.v, false, vb0.v,
                                                    (short)0, A, false, false);
        A = __builtin_amdgcn_wmma_f32_16x16x32_bf16(false, pa1.v, false, vb1.v,
                                                    (short)0, A, false, false);
        acc[t] = A;
      }
    }
  }

  // ---- epilogue: normalize & store Z ----
  #pragma unroll
  for (int t = 0; t < 13; ++t) {
    const int tt = wid + 4 * t;
    if (tt < 49) {
      #pragma unroll
      for (int i = 0; i < 8; ++i) {
        const int row = i + 8 * half;
        const int q   = qt * 16 + row;
        Z[((size_t)b * NQ + q) * DQ + tt * 16 + l16] = acc[t][i] / lrun[i];
      }
    }
  }
}

// ---------------------------------------------------------------------------
// Kernel 3: fold (overlap-add / counts), 1x1 conv back to 64ch, residual.
// ---------------------------------------------------------------------------
__global__ void fold_conv_kernel(const float* __restrict__ x,
                                 const float* __restrict__ Z,
                                 const float* __restrict__ ow,
                                 const float* __restrict__ ob,
                                 float* __restrict__ out)
{
  const int tid = blockIdx.x * 256 + threadIdx.x;
  if (tid >= B_ * HH * WW) return;
  const int xc = tid & 127;
  const int y  = (tid >> 7) & 127;
  const int b  = tid >> 14;

  const int yy = y + PADP, xx = xc + PADP;       // padded coords (>=3)
  const int qy_lo = (yy - 3) >> 2;
  const int qy_hi = imin(31, yy >> 2);
  const int qx_lo = (xx - 3) >> 2;
  const int qx_hi = imin(31, xx >> 2);

  float y16[16];
  #pragma unroll
  for (int c = 0; c < 16; ++c) y16[c] = 0.0f;
  int cnt = 0;
  for (int qy = qy_lo; qy <= qy_hi; ++qy) {
    for (int qx = qx_lo; qx <= qx_hi; ++qx) {
      const int ph = yy - qy * 4, pw = xx - qx * 4;
      const int p  = ph * 7 + pw;
      const float* z = Z + ((size_t)b * NQ + qy * 32 + qx) * DQ + p * 16;
      #pragma unroll
      for (int c = 0; c < 16; ++c) y16[c] += z[c];
      ++cnt;
    }
  }
  const float inv = 1.0f / (float)cnt;
  #pragma unroll
  for (int c = 0; c < 16; ++c) y16[c] *= inv;

  const size_t po = (size_t)b * CIN * (HH * WW) + (size_t)y * WW + xc;
  for (int co = 0; co < CIN; ++co) {
    float r = ob[co];
    #pragma unroll
    for (int c = 0; c < 16; ++c) r += ow[co * 16 + c] * y16[c];
    out[po + (size_t)co * (HH * WW)] = x[po + (size_t)co * (HH * WW)] + r;
  }
}

// ---------------------------------------------------------------------------
extern "C" void kernel_launch(void* const* d_in, const int* in_sizes, int n_in,
                              void* d_out, int out_size, void* d_ws, size_t ws_size,
                              hipStream_t stream)
{
  (void)in_sizes; (void)n_in; (void)out_size; (void)ws_size;

  const float* bx   = (const float*)d_in[0];
  const float* g_w  = (const float*)d_in[1];
  const float* g_b  = (const float*)d_in[2];
  const float* th_w = (const float*)d_in[3];
  const float* th_b = (const float*)d_in[4];
  const float* ph_w = (const float*)d_in[5];
  const float* ph_b = (const float*)d_in[6];
  const float* o_w  = (const float*)d_in[7];
  const float* o_b  = (const float*)d_in[8];
  float* out = (float*)d_out;

  char* ws = (char*)d_ws;
  const size_t szQ = (size_t)B_ * HP * WP * 16 * sizeof(__bf16);   // 2,298,368 B
  const size_t szV = (size_t)B_ * 16 * HP * WPV * sizeof(__bf16);  // 2,469,888 B
  __bf16* o1 = (__bf16*)(ws);                    // queries (g), ch-last
  __bf16* o3 = (__bf16*)(ws + szQ);              // keys (ph), ch-last
  __bf16* o2 = (__bf16*)(ws + 2 * szQ);          // values (th), ch-first padded
  float*  Z  = (float*)(ws + 2 * szQ + szV);     // [B][1024][784] fp32

  // 1) QKV pack: B*HP*WP*48 threads
  {
    const long long total = (long long)B_ * HP * WP * 48;  // 3,447,552
    const int blocks = (int)((total + 255) / 256);
    qkv_pack_kernel<<<blocks, 256, 0, stream>>>(bx, g_w, g_b, th_w, th_b,
                                                ph_w, ph_b, o1, o3, o2);
  }
  // 2) Flash attention: one WG per 16-query tile (4 batches * 64 tiles)
  attn_kernel<<<256, 128, 0, stream>>>(o1, o3, o2, Z);

  // 3) Fold + counts + 1x1 conv + residual
  fold_conv_kernel<<<256, 256, 0, stream>>>(bx, Z, o_w, o_b, out);
}
